// C_MFN_30030411333980
// MI455X (gfx1250) — matrix operand order repaired and checked by reading, not compiled, not run on hardware
//
#include <hip/hip_runtime.h>
#include <stdint.h>

typedef __bf16 bf16;
typedef __attribute__((ext_vector_type(16))) __bf16 v16bf;
typedef __attribute__((ext_vector_type(8)))  __bf16 v8bf;
typedef __attribute__((ext_vector_type(8)))  float   v8f;

// ---- problem constants ----
#define NB     512         // batch
#define XW     752         // T_IN+A_IN+V_IN
#define CSTARD 1024        // 2*(T_H+A_H+V_H)
#define MEMD   512
#define BOTHD  1536        // CSTARD + MEM
#define NSTEP  128
// concat [x_pad | h] K dims (all multiples of 32)
#define KT 576             // 320 (x pad of 300) + 256 (h)
#define KA 224             // 96  (x pad of 81)  + 128 (h)
#define KV 512             // 384 (x pad of 371) + 128 (h)

#define ACT_NONE 0
#define ACT_RELU 1
#define ACT_SIG  2
#define ACT_TANH 3

// ===========================================================================
// Templated NT GEMM:  out = ACT( A[M,K]_bf16 @ W[N,K]_bf16^T + bias[N] )
// A row-major (lda), W row-major (ldw): both K-contiguous, so the CDNA5
// 16x16x32 bf16 WMMA A/B fragments are plain contiguous 16B loads.
// Block: 256 thr = 8 waves, waves 2(M) x 4(N), each wave 32x64 => tile 64x256.
// Requires M%64==0, N%256==0, K%32==0.  WC: write f32 C, WCB: write bf16 Cb.
// ===========================================================================
template<int ACT, bool WC, bool WCB>
__global__ __launch_bounds__(256)
void gemm_bf16_nt(const bf16* __restrict__ A, int lda,
                  const bf16* __restrict__ W, int ldw,
                  const float* __restrict__ bias,
                  float* __restrict__ C, int ldc,
                  bf16* __restrict__ Cb, int ldcb,
                  int M, int N, int K)
{
    const int wave = threadIdx.x >> 5;
    const int lane = threadIdx.x & 31;
    const int wm = wave >> 2, wn = wave & 3;
    const int m0 = blockIdx.y * 64 + wm * 32;
    const int n0 = blockIdx.x * 256 + wn * 64;
    if (m0 >= M || n0 >= N) return;
    const int hf  = lane >> 4;   // K-half select (ISA wave32 A/B layouts)
    const int l15 = lane & 15;

    v8f acc[2][4] = {};

    for (int k0 = 0; k0 < K; k0 += 32) {
        if (k0 + 32 < K) {  // -> global_prefetch_b8 for next K tile
            __builtin_prefetch(A + (size_t)(m0 + l15) * lda + k0 + 32, 0, 1);
            __builtin_prefetch(W + (size_t)(n0 + l15) * ldw + k0 + 32, 0, 1);
        }
        // A frag (16x32): lane row m=l15; elems 0..7 -> K=k0+hf*8+e,
        //                 elems 8..15 -> K=k0+16+hf*8+(e-8)
        v16bf afrag[2];
#pragma unroll
        for (int i = 0; i < 2; i++) {
            const bf16* ap = A + (size_t)(m0 + i * 16 + l15) * lda + (k0 + hf * 8);
            v8bf lo = *(const v8bf*)(ap);
            v8bf hi = *(const v8bf*)(ap + 16);
            v16bf t;
#pragma unroll
            for (int e = 0; e < 8; e++) { t[e] = lo[e]; t[e + 8] = hi[e]; }
            afrag[i] = t;
        }
        // B frag (32x16): lane col n=l15; elems 0..7 -> K=k0+hf*8+e, elems 8..15 -> K=k0+16+hf*8+(e-8);
        // B[k][n] = W[n][k] => contiguous run of the weight row.
        v16bf bfrag[4];
#pragma unroll
        for (int j = 0; j < 4; j++) {
            const bf16* bp = W + (size_t)(n0 + j * 16 + l15) * ldw + (k0 + hf * 8);
            v8bf lo = *(const v8bf*)(bp);
            v8bf hi = *(const v8bf*)(bp + 16);
            v16bf t;
#pragma unroll
            for (int e = 0; e < 8; e++) { t[e] = lo[e]; t[e + 8] = hi[e]; }
            bfrag[j] = t;
        }
#pragma unroll
        for (int i = 0; i < 2; i++)
#pragma unroll
            for (int j = 0; j < 4; j++)
                acc[i][j] = __builtin_amdgcn_wmma_f32_16x16x32_bf16(
                    false, afrag[i], false, bfrag[j], (short)0, acc[i][j],
                    false, false);
    }

    // C/D layout: lane col n=l15, rows = hf*8 + r (r = vgpr index)
#pragma unroll
    for (int j = 0; j < 4; j++) {
        const int nn = n0 + j * 16 + l15;
        const float bv = bias[nn];           // hoisted: one load per col-tile
#pragma unroll
        for (int i = 0; i < 2; i++) {
#pragma unroll
            for (int r = 0; r < 8; r++) {
                const int mm = m0 + i * 16 + hf * 8 + r;
                float v = acc[i][j][r] + bv;
                if (ACT == ACT_RELU) v = v > 0.f ? v : 0.f;
                if (ACT == ACT_SIG)  v = 1.f / (1.f + expf(-v));
                if (ACT == ACT_TANH) v = tanhf(v);
                if (WC)  C[(size_t)mm * ldc + nn] = v;
                if (WCB) Cb[(size_t)mm * ldcb + nn] = (bf16)v;
            }
        }
    }
}

// ---- pack / convert helpers ---------------------------------------------
// pack f32 weight block into bf16 at dst[r*dld + dcol0 + c], c in [0,wpad),
// zero-padding cols >= cols. Used to build concatenated [Wih|Whh] weights.
__global__ void k_pack_w(const float* __restrict__ s, int rows, int cols,
                         bf16* __restrict__ d, int dld, int dcol0,
                         int wpad, int total) {
    int i = blockIdx.x * 256 + threadIdx.x;
    if (i >= total) return;
    int r = i / wpad, c = i - r * wpad;
    d[(size_t)r * dld + dcol0 + c] =
        (c < cols) ? (bf16)s[(size_t)r * cols + c] : (bf16)0.f;
}
// split one timestep of x_p into the three [x_pad | h] concat buffers
__global__ void k_pack_x_all(const float* __restrict__ x,
                             bf16* __restrict__ xh_t, bf16* __restrict__ xh_a,
                             bf16* __restrict__ xh_v, int total) {  // NB*800
    int i = blockIdx.x * 256 + threadIdx.x;
    if (i >= total) return;
    int row = i / 800, c = i - row * 800;
    const float* xr = x + (size_t)row * XW;
    if (c < 320)      xh_t[(size_t)row * KT + c]       = (c < 300) ? (bf16)xr[c]           : (bf16)0.f;
    else if (c < 416) { int k = c - 320;
                        xh_a[(size_t)row * KA + k]      = (k < 81)  ? (bf16)xr[300 + k]     : (bf16)0.f; }
    else              { int k = c - 416;
                        xh_v[(size_t)row * KV + k]      = (k < 371) ? (bf16)xr[381 + k]     : (bf16)0.f; }
}
__global__ void k_add2(const float* a, const float* b, float* o, int n) {
    int i = blockIdx.x * 256 + threadIdx.x;
    if (i < n) o[i] = a[i] + b[i];
}
__global__ void k_copyf(const float* s, float* d, int n) {
    int i = blockIdx.x * 256 + threadIdx.x;
    if (i < n) d[i] = s[i];
}
__global__ void k_fillbf(bf16* d, int n) {
    int i = blockIdx.x * 256 + threadIdx.x;
    if (i < n) d[i] = (bf16)0.f;
}
__global__ void k_initm(const float* __restrict__ mem, float* __restrict__ m,
                        bf16* __restrict__ both_mem, int total) {
    int i = blockIdx.x * 256 + threadIdx.x;
    if (i >= total) return;
    int r = i >> 9, c = i & 511;
    float v = mem[i];
    m[i] = v;
    both_mem[(size_t)r * BOTHD + c] = (bf16)v;
}

// ---- fused pointwise stages ---------------------------------------------
// all three LSTM cells in one launch; h (bf16) written into concat buffers
__global__ void k_lstm_all(const float* __restrict__ gt, const float* __restrict__ ga,
                           const float* __restrict__ gv,
                           const float* __restrict__ ctp, const float* __restrict__ cap,
                           const float* __restrict__ cvp,
                           float* __restrict__ ctn, float* __restrict__ can,
                           float* __restrict__ cvn,
                           bf16* __restrict__ xh_t, bf16* __restrict__ xh_a,
                           bf16* __restrict__ xh_v, int total) {  // NB*512
    int i = blockIdx.x * 256 + threadIdx.x;
    if (i >= total) return;
    int row = i >> 9, c = i & 511;
    const float *g, *cp; float* cn; bf16* h; int H, k, ld, off;
    if (c < 256)      { g = gt; cp = ctp; cn = ctn; h = xh_t; H = 256; k = c;       ld = KT; off = 320; }
    else if (c < 384) { g = ga; cp = cap; cn = can; h = xh_a; H = 128; k = c - 256; ld = KA; off = 96;  }
    else              { g = gv; cp = cvp; cn = cvn; h = xh_v; H = 128; k = c - 384; ld = KV; off = 384; }
    const float* gr = g + (size_t)row * 4 * H;
    float ig = 1.f / (1.f + expf(-gr[k]));
    float fg = 1.f / (1.f + expf(-gr[H + k]));
    float gg = tanhf(gr[2 * H + k]);
    float og = 1.f / (1.f + expf(-gr[3 * H + k]));
    float cc = fg * cp[(size_t)row * H + k] + ig * gg;
    cn[(size_t)row * H + k] = cc;
    h[(size_t)row * ld + off + k] = (bf16)(og * tanhf(cc));
}

__global__ void k_build_cstar(const float* __restrict__ pt, const float* __restrict__ pa,
                              const float* __restrict__ pv, const float* __restrict__ nt,
                              const float* __restrict__ na, const float* __restrict__ nv,
                              float* __restrict__ cs, bf16* __restrict__ csb, int total) {
    int i = blockIdx.x * 256 + threadIdx.x;
    if (i >= total) return;
    int r = i >> 10, c = i & 1023;
    float v;
    if      (c < 256) v = pt[(size_t)r * 256 + c];
    else if (c < 384) v = pa[(size_t)r * 128 + c - 256];
    else if (c < 512) v = pv[(size_t)r * 128 + c - 384];
    else if (c < 768) v = nt[(size_t)r * 256 + c - 512];
    else if (c < 896) v = na[(size_t)r * 128 + c - 768];
    else              v = nv[(size_t)r * 128 + c - 896];
    cs[i] = v;
    csb[i] = (bf16)v;
}

// softmax over 1024 features, then attended = attn * c_star -> both[:, :1024]
__global__ void k_softmax_attend(const float* __restrict__ logits,
                                 const float* __restrict__ cstar,
                                 bf16* __restrict__ both) {
    __shared__ float red[256];
    const int row = blockIdx.x, t = threadIdx.x;
    const float* lr = logits + (size_t)row * CSTARD;
    float mx = -3.4e38f;
    for (int k = t; k < CSTARD; k += 256) mx = fmaxf(mx, lr[k]);
    red[t] = mx; __syncthreads();
    for (int s = 128; s > 0; s >>= 1) { if (t < s) red[t] = fmaxf(red[t], red[t + s]); __syncthreads(); }
    mx = red[0]; __syncthreads();
    float sum = 0.f;
    for (int k = t; k < CSTARD; k += 256) sum += expf(lr[k] - mx);
    red[t] = sum; __syncthreads();
    for (int s = 128; s > 0; s >>= 1) { if (t < s) red[t] += red[t + s]; __syncthreads(); }
    const float inv = 1.f / red[0];
    for (int k = t; k < CSTARD; k += 256) {
        float a = expf(lr[k] - mx) * inv;
        both[(size_t)row * BOTHD + k] = (bf16)(a * cstar[(size_t)row * CSTARD + k]);
    }
}

// m = g1*m + g2*chat ; refresh bf16 copy inside `both`
__global__ void k_mem_update(const float* __restrict__ g1, const float* __restrict__ g2,
                             const float* __restrict__ chat, float* __restrict__ m,
                             bf16* __restrict__ both_mem, int total) {
    int i = blockIdx.x * 256 + threadIdx.x;
    if (i >= total) return;
    int r = i >> 9, c = i & 511;
    float nm = g1[i] * m[i] + g2[i] * chat[i];
    m[i] = nm;
    both_mem[(size_t)r * BOTHD + c] = (bf16)nm;
}

// last_h = [h_t | h_a | h_v | m]  (h_* are strided views into concat buffers)
__global__ void k_build_last(const bf16* __restrict__ ht, const bf16* __restrict__ ha,
                             const bf16* __restrict__ hv, const float* __restrict__ m,
                             bf16* __restrict__ last, int total) {
    int i = blockIdx.x * 256 + threadIdx.x;
    if (i >= total) return;
    int r = i >> 10, c = i & 1023;
    bf16 v;
    if      (c < 256) v = ht[(size_t)r * KT + c];
    else if (c < 384) v = ha[(size_t)r * KA + c - 256];
    else if (c < 512) v = hv[(size_t)r * KV + c - 384];
    else              v = (bf16)m[(size_t)r * 512 + c - 512];
    last[i] = v;
}

// out[row] = dot(h1[row,:256], W2[0,:256]) + b2   (OUT_D == 1)
__global__ void k_final_out(const bf16* __restrict__ h1, const float* __restrict__ W2,
                            const float* __restrict__ b2, float* __restrict__ out) {
    __shared__ float red[256];
    const int row = blockIdx.x, t = threadIdx.x;
    red[t] = (float)h1[(size_t)row * 256 + t] * W2[t];
    __syncthreads();
    for (int s = 128; s > 0; s >>= 1) { if (t < s) red[t] += red[t + s]; __syncthreads(); }
    if (t == 0) out[row] = red[0] + b2[0];
}

// ===========================================================================
extern "C" void kernel_launch(void* const* d_in, const int* in_sizes, int n_in,
                              void* d_out, int out_size, void* d_ws, size_t ws_size,
                              hipStream_t stream)
{
    (void)in_sizes; (void)n_in; (void)out_size; (void)ws_size;
    const float* x_p     = (const float*)d_in[0];
    const float* c_t_in  = (const float*)d_in[1];
    const float* c_a_in  = (const float*)d_in[2];
    const float* c_v_in  = (const float*)d_in[3];
    const float* mem_in  = (const float*)d_in[4];
    const float* t_Wih = (const float*)d_in[5];  const float* t_Whh = (const float*)d_in[6];
    const float* t_bih = (const float*)d_in[7];  const float* t_bhh = (const float*)d_in[8];
    const float* a_Wih = (const float*)d_in[9];  const float* a_Whh = (const float*)d_in[10];
    const float* a_bih = (const float*)d_in[11]; const float* a_bhh = (const float*)d_in[12];
    const float* v_Wih = (const float*)d_in[13]; const float* v_Whh = (const float*)d_in[14];
    const float* v_bih = (const float*)d_in[15]; const float* v_bhh = (const float*)d_in[16];
    const float* a1W1 = (const float*)d_in[17]; const float* a1b1 = (const float*)d_in[18];
    const float* a1W2 = (const float*)d_in[19]; const float* a1b2 = (const float*)d_in[20];
    const float* a2W1 = (const float*)d_in[21]; const float* a2b1 = (const float*)d_in[22];
    const float* a2W2 = (const float*)d_in[23]; const float* a2b2 = (const float*)d_in[24];
    const float* g1W1 = (const float*)d_in[25]; const float* g1b1 = (const float*)d_in[26];
    const float* g1W2 = (const float*)d_in[27]; const float* g1b2 = (const float*)d_in[28];
    const float* g2W1 = (const float*)d_in[29]; const float* g2b1 = (const float*)d_in[30];
    const float* g2W2 = (const float*)d_in[31]; const float* g2b2 = (const float*)d_in[32];
    const float* oW1  = (const float*)d_in[33]; const float* ob1  = (const float*)d_in[34];
    const float* oW2  = (const float*)d_in[35]; const float* ob2  = (const float*)d_in[36];

    // ---- workspace layout (256B aligned slabs) ----
    uintptr_t cur = (uintptr_t)d_ws;
    auto alloc = [&](size_t bytes) -> void* {
        uintptr_t r = (cur + 255) & ~(uintptr_t)255; cur = r + bytes; return (void*)r;
    };
    auto abf = [&](size_t n) { return (bf16*)alloc(n * sizeof(bf16)); };
    auto af  = [&](size_t n) { return (float*)alloc(n * sizeof(float)); };

    // bf16 weights (concatenated [Wih|Whh] for the LSTMs, K zero-padded)
    bf16 *Wcat_t = abf(1024 * KT), *Wcat_a = abf(512 * KA), *Wcat_v = abf(512 * KV);
    bf16 *A1W1 = abf(512 * 1024),  *A1W2 = abf(1024 * 512);
    bf16 *A2W1 = abf(512 * 1024),  *A2W2 = abf(512 * 512);
    bf16 *G1W1 = abf(512 * 1536),  *G1W2 = abf(512 * 512);
    bf16 *G2W1 = abf(512 * 1536),  *G2W2 = abf(512 * 512);
    bf16 *OW1  = abf(256 * 1024);
    float *bs_t = af(1024), *bs_a = af(512), *bs_v = af(512);
    // activations
    bf16 *xh_t = abf(NB * KT), *xh_a = abf(NB * KA), *xh_v = abf(NB * KV);
    float *gates_t = af(NB * 1024), *gates_a = af(NB * 512), *gates_v = af(NB * 512);
    float *ctb = af(2 * NB * 256), *cab = af(2 * NB * 128), *cvb = af(2 * NB * 128);
    float *ct[2] = { ctb, ctb + NB * 256 };
    float *ca[2] = { cab, cab + NB * 128 };
    float *cv[2] = { cvb, cvb + NB * 128 };
    float *cstar = af(NB * CSTARD);  bf16 *cstarb = abf(NB * CSTARD);
    bf16 *h1b = abf(NB * 512);       float *logits = af(NB * CSTARD);
    bf16 *both = abf(NB * BOTHD);    bf16 *h2b = abf(NB * 512);
    float *chat = af(NB * 512);
    bf16 *gh1b = abf(NB * 512), *gh2b = abf(NB * 512);
    float *gamma1 = af(NB * 512), *gamma2 = af(NB * 512);
    float *m = af(NB * MEMD);
    bf16 *lastb = abf(NB * 1024), *o1b = abf(NB * 256);

    auto NBLK = [](int n) { return (n + 255) / 256; };
    auto packw = [&](const float* s, int r, int c, bf16* d, int dld, int dcol0, int wpad) {
        k_pack_w<<<NBLK(r * wpad), 256, 0, stream>>>(s, r, c, d, dld, dcol0, wpad, r * wpad);
    };
    auto gemm = [&](const bf16* A, int lda, const bf16* W, int ldw, const float* bias,
                    float* C, int ldc, bf16* Cb, int ldcb, int M, int N, int K, int act) {
        dim3 g(N / 256, M / 64);
        if (act == ACT_RELU)
            gemm_bf16_nt<ACT_RELU, false, true ><<<g, 256, 0, stream>>>(A, lda, W, ldw, bias, C, ldc, Cb, ldcb, M, N, K);
        else if (act == ACT_TANH)
            gemm_bf16_nt<ACT_TANH, true,  false><<<g, 256, 0, stream>>>(A, lda, W, ldw, bias, C, ldc, Cb, ldcb, M, N, K);
        else if (act == ACT_SIG)
            gemm_bf16_nt<ACT_SIG,  true,  false><<<g, 256, 0, stream>>>(A, lda, W, ldw, bias, C, ldc, Cb, ldcb, M, N, K);
        else
            gemm_bf16_nt<ACT_NONE, true,  false><<<g, 256, 0, stream>>>(A, lda, W, ldw, bias, C, ldc, Cb, ldcb, M, N, K);
    };

    // ---- one-time (per launch) prep ----
    packw(t_Wih, 1024, 300, Wcat_t, KT, 0,   320);
    packw(t_Whh, 1024, 256, Wcat_t, KT, 320, 256);
    packw(a_Wih, 512,  81,  Wcat_a, KA, 0,   96);
    packw(a_Whh, 512,  128, Wcat_a, KA, 96,  128);
    packw(v_Wih, 512,  371, Wcat_v, KV, 0,   384);
    packw(v_Whh, 512,  128, Wcat_v, KV, 384, 128);
    packw(a1W1, 512, 1024, A1W1, 1024, 0, 1024);  packw(a1W2, 1024, 512, A1W2, 512, 0, 512);
    packw(a2W1, 512, 1024, A2W1, 1024, 0, 1024);  packw(a2W2, 512,  512, A2W2, 512, 0, 512);
    packw(g1W1, 512, 1536, G1W1, 1536, 0, 1536);  packw(g1W2, 512,  512, G1W2, 512, 0, 512);
    packw(g2W1, 512, 1536, G2W1, 1536, 0, 1536);  packw(g2W2, 512,  512, G2W2, 512, 0, 512);
    packw(oW1, 256, 1024, OW1, 1024, 0, 1024);
    k_add2<<<NBLK(1024), 256, 0, stream>>>(t_bih, t_bhh, bs_t, 1024);
    k_add2<<<NBLK(512), 256, 0, stream>>>(a_bih, a_bhh, bs_a, 512);
    k_add2<<<NBLK(512), 256, 0, stream>>>(v_bih, v_bhh, bs_v, 512);
    k_copyf<<<NBLK(NB * 256), 256, 0, stream>>>(c_t_in, ct[0], NB * 256);
    k_copyf<<<NBLK(NB * 128), 256, 0, stream>>>(c_a_in, ca[0], NB * 128);
    k_copyf<<<NBLK(NB * 128), 256, 0, stream>>>(c_v_in, cv[0], NB * 128);
    k_initm<<<NBLK(NB * 512), 256, 0, stream>>>(mem_in, m, both + 1024, NB * 512);
    k_fillbf<<<NBLK(NB * KT), 256, 0, stream>>>(xh_t, NB * KT);   // h(0)=0 + pad
    k_fillbf<<<NBLK(NB * KA), 256, 0, stream>>>(xh_a, NB * KA);
    k_fillbf<<<NBLK(NB * KV), 256, 0, stream>>>(xh_v, NB * KV);

    // ---- recurrence (128 dependent steps; everything L2-resident) ----
    int p = 0;
    for (int t = 0; t < NSTEP; t++) {
        const int q = p ^ 1;
        const float* xs = x_p + (size_t)t * NB * XW;
        k_pack_x_all<<<NBLK(NB * 800), 256, 0, stream>>>(xs, xh_t, xh_a, xh_v, NB * 800);
        // one fused GEMM per LSTM: gates = [x|h] @ [Wih|Whh]^T + (bih+bhh)
        gemm(xh_t, KT, Wcat_t, KT, bs_t, gates_t, 1024, nullptr, 0, NB, 1024, KT, ACT_NONE);
        gemm(xh_a, KA, Wcat_a, KA, bs_a, gates_a, 512,  nullptr, 0, NB, 512,  KA, ACT_NONE);
        gemm(xh_v, KV, Wcat_v, KV, bs_v, gates_v, 512,  nullptr, 0, NB, 512,  KV, ACT_NONE);
        k_lstm_all<<<NBLK(NB * 512), 256, 0, stream>>>(gates_t, gates_a, gates_v,
                                                       ct[p], ca[p], cv[p],
                                                       ct[q], ca[q], cv[q],
                                                       xh_t, xh_a, xh_v, NB * 512);
        k_build_cstar<<<NBLK(NB * CSTARD), 256, 0, stream>>>(ct[p], ca[p], cv[p],
                                                             ct[q], ca[q], cv[q],
                                                             cstar, cstarb, NB * CSTARD);
        // attention
        gemm(cstarb, 1024, A1W1, 1024, a1b1, nullptr, 0, h1b, 512, NB, 512, 1024, ACT_RELU);
        gemm(h1b, 512, A1W2, 512, a1b2, logits, 1024, nullptr, 0, NB, 1024, 512, ACT_NONE);
        k_softmax_attend<<<NB, 256, 0, stream>>>(logits, cstar, both);
        // c_hat (tanh fused into epilogue)
        gemm(both, BOTHD, A2W1, 1024, a2b1, nullptr, 0, h2b, 512, NB, 512, 1024, ACT_RELU);
        gemm(h2b, 512, A2W2, 512, a2b2, chat, 512, nullptr, 0, NB, 512, 512, ACT_TANH);
        // gamma1 / gamma2 (sigmoid fused)
        gemm(both, BOTHD, G1W1, 1536, g1b1, nullptr, 0, gh1b, 512, NB, 512, 1536, ACT_RELU);
        gemm(gh1b, 512, G1W2, 512, g1b2, gamma1, 512, nullptr, 0, NB, 512, 512, ACT_SIG);
        gemm(both, BOTHD, G2W1, 1536, g2b1, nullptr, 0, gh2b, 512, NB, 512, 1536, ACT_RELU);
        gemm(gh2b, 512, G2W2, 512, g2b2, gamma2, 512, nullptr, 0, NB, 512, 512, ACT_SIG);
        k_mem_update<<<NBLK(NB * 512), 256, 0, stream>>>(gamma1, gamma2, chat, m,
                                                         both + 1024, NB * 512);
        p = q;
    }

    // ---- output head ----
    k_build_last<<<NBLK(NB * 1024), 256, 0, stream>>>(xh_t + 320, xh_a + 96, xh_v + 384,
                                                      m, lastb, NB * 1024);
    gemm(lastb, 1024, OW1, 1024, ob1, nullptr, 0, o1b, 256, NB, 256, 1024, ACT_RELU);
    k_final_out<<<NB, 256, 0, stream>>>(o1b, oW2, ob2, (float*)d_out);
}
